// CrossAttention_72121090834620
// MI455X (gfx1250) — matrix operand order repaired and checked by reading, not compiled, not run on hardware
//
#include <hip/hip_runtime.h>
#include <math.h>

// CDNA5 (gfx1250) wave32 WMMA types
typedef __attribute__((ext_vector_type(16))) _Float16 v16h;
typedef __attribute__((ext_vector_type(8)))  float    v8f;

union V16H { v16h v; _Float16 h[16]; unsigned u[8]; };
union V8F  { v8f  v; float f[8]; };
union H8   { _Float16 h[8]; uint4 u4; };

#define WMMA_F16(a, b, c) \
    __builtin_amdgcn_wmma_f32_16x16x32_f16(false, (a), false, (b), (short)0, (c), false, false)

static constexpr float SCALE = 0.03125f;  // 1024^-0.5

// LDS byte address = low 32 bits of the generic pointer (AS3->generic cast
// places the LDS offset in the low dword; flat aperture truncates to [31:0]).
__device__ __forceinline__ unsigned lds_addr32(const void* p) {
    return (unsigned)(unsigned long long)p;
}

// ---------------------------------------------------------------------------
// Kernel A: O[m,n] = sum_k A[m,k] * W[n,k]   (NT GEMM, fp32 in, f16 out,
// f32 accumulate). Block = 256 thr = 8 waves, block tile 64x128,
// per-wave 32x32 (4 WMMA frags, 2x A/B fragment reuse), K-step 32.
// ---------------------------------------------------------------------------
__global__ __launch_bounds__(256) void k_linear_nt(
    const float* __restrict__ A,      // M x K row-major (M=320, K=4096)
    const float* __restrict__ W,      // N x K row-major (N=4096)
    _Float16*    __restrict__ O,      // M x N row-major, f16
    int M, int N, int K)
{
    __shared__ _Float16 sA[64][40];    // 5120 B  (padded stride 80 B)
    __shared__ _Float16 sW[128][40];   // 10240 B

    const int tid = threadIdx.x;
    const int m0  = blockIdx.x * 64;
    const int n0  = blockIdx.y * 128;

    // staging: A tile 64x32 -> 8 floats/thread; W tile 128x32 -> 16 floats/thread
    const int arow = tid >> 2;              // 0..63
    const int akk  = (tid & 3) * 8;         // 0,8,16,24
    const int wrow = tid >> 1;              // 0..127
    const int wkk  = (tid & 1) * 16;        // 0,16
    const float* gA = A + (size_t)(m0 + arow) * K + akk;
    const float* gW = W + (size_t)(n0 + wrow) * K + wkk;

    const int lane = tid & 31;
    const int wave = tid >> 5;
    const int wm = wave & 1;                // 0..1 -> 32-row stripe
    const int wn = wave >> 1;               // 0..3 -> 32-col stripe
    const int l16 = lane & 15, lh = lane >> 4;

    V8F acc[2][2];
    #pragma unroll
    for (int t = 0; t < 2; ++t)
        #pragma unroll
        for (int u = 0; u < 2; ++u)
            #pragma unroll
            for (int r = 0; r < 8; ++r) acc[t][u].f[r] = 0.0f;

    for (int k0 = 0; k0 < K; k0 += 32) {
        if (k0 + 32 < K) {                          // gfx1250 global_prefetch_b8
            __builtin_prefetch(gA + k0 + 32, 0, 1);
            __builtin_prefetch(gW + k0 + 32, 0, 1);
        }
        // ---- stage fp32 -> f16 into LDS ----
        {
            const float4 a0 = *reinterpret_cast<const float4*>(gA + k0);
            const float4 a1 = *reinterpret_cast<const float4*>(gA + k0 + 4);
            H8 pa;
            pa.h[0]=(_Float16)a0.x; pa.h[1]=(_Float16)a0.y;
            pa.h[2]=(_Float16)a0.z; pa.h[3]=(_Float16)a0.w;
            pa.h[4]=(_Float16)a1.x; pa.h[5]=(_Float16)a1.y;
            pa.h[6]=(_Float16)a1.z; pa.h[7]=(_Float16)a1.w;
            *reinterpret_cast<uint4*>(&sA[arow][akk]) = pa.u4;

            #pragma unroll
            for (int half8 = 0; half8 < 2; ++half8) {
                const float4 w0 = *reinterpret_cast<const float4*>(gW + k0 + half8 * 8);
                const float4 w1 = *reinterpret_cast<const float4*>(gW + k0 + half8 * 8 + 4);
                H8 pw;
                pw.h[0]=(_Float16)w0.x; pw.h[1]=(_Float16)w0.y;
                pw.h[2]=(_Float16)w0.z; pw.h[3]=(_Float16)w0.w;
                pw.h[4]=(_Float16)w1.x; pw.h[5]=(_Float16)w1.y;
                pw.h[6]=(_Float16)w1.z; pw.h[7]=(_Float16)w1.w;
                *reinterpret_cast<uint4*>(&sW[wrow][wkk + half8 * 8]) = pw.u4;
            }
        }
        __syncthreads();

        // ---- B fragments (32x16 f16): lane = column n, K split by lane half ----
        V16H bf[2];
        #pragma unroll
        for (int u = 0; u < 2; ++u) {
            const int nn = wn * 32 + u * 16 + l16;
            #pragma unroll
            for (int g = 0; g < 8; ++g)
                bf[u].u[g] = *reinterpret_cast<const unsigned*>(&sW[nn][((g & 3) * 2) + ((g >> 2) * 16) + lh * 8]);
        }

        // ---- A fragments (16x32 f16) + 4 WMMA ----
        #pragma unroll
        for (int t = 0; t < 2; ++t) {
            V16H af;
            const int mm = wm * 32 + t * 16 + l16;
            #pragma unroll
            for (int g = 0; g < 8; ++g) {
                const int kf = ((g & 3) * 2) + ((g >> 2) * 16) + lh * 8;
                af.u[g] = *reinterpret_cast<const unsigned*>(&sA[mm][kf]);
            }
            #pragma unroll
            for (int u = 0; u < 2; ++u)
                acc[t][u].v = WMMA_F16(af.v, bf[u].v, acc[t][u].v);
        }
        __syncthreads();
    }

    // ---- epilogue: f32 acc -> f16 (C/D layout: n=lane%16, m=r+8*(lane/16)) ----
    #pragma unroll
    for (int t = 0; t < 2; ++t)
        #pragma unroll
        for (int u = 0; u < 2; ++u) {
            const int mg = m0 + wm * 32 + t * 16 + lh * 8;
            const int ng = n0 + wn * 32 + u * 16 + l16;
            #pragma unroll
            for (int r = 0; r < 8; ++r)
                O[(size_t)(mg + r) * N + ng] = (_Float16)acc[t][u].f[r];
        }
}

// ---------------------------------------------------------------------------
// Kernel B: kkt[b,h,i,j] = SCALE * sum_d k[b,h,i,d]*k[b,h,j,d]
// One wave per 16x16 tile, K=1024. Tiles staged into LDS with the gfx1250
// async-copy path (global_load_async_to_lds_b128 + s_wait_asynccnt), then
// WMMA fragments read from LDS.
// ---------------------------------------------------------------------------
__global__ __launch_bounds__(32) void k_kkt(
    const _Float16* __restrict__ Kf,   // (320, 4096) f16
    _Float16*       __restrict__ KKT)  // (8, 160, 160) f16, pre-scaled
{
    // 16 rows x 32 halfs per tile; row stride padded to 48 halfs (96 B, 16B-aligned)
    __shared__ _Float16 sI[16 * 48];
    __shared__ _Float16 sJ[16 * 48];

    const int bh = blockIdx.z;
    const int b  = bh >> 2;
    const int h  = bh & 3;
    const int i0 = blockIdx.y * 16, j0 = blockIdx.x * 16;
    const int lane = threadIdx.x;
    const int l16 = lane & 15, lh = lane >> 4;

    const _Float16* base = Kf + (size_t)b * 160 * 4096 + (size_t)h * 1024;

    V8F acc;
    #pragma unroll
    for (int r = 0; r < 8; ++r) acc.f[r] = 0.0f;

    for (int k0 = 0; k0 < 1024; k0 += 32) {
        // ---- async stage: 64 chunks of 16 B per tile, 2 chunks/lane ----
        #pragma unroll
        for (int c = 0; c < 2; ++c) {
            const int chunk = lane + c * 32;         // 0..63
            const int rrow  = chunk >> 2;            // 0..15
            const int part  = chunk & 3;             // 16B sub-chunk in row
            const unsigned long long ga = (unsigned long long)
                (base + (size_t)(i0 + rrow) * 4096 + k0 + part * 8);
            const unsigned long long gb = (unsigned long long)
                (base + (size_t)(j0 + rrow) * 4096 + k0 + part * 8);
            const unsigned la = lds_addr32(&sI[rrow * 48 + part * 8]);
            const unsigned lb = lds_addr32(&sJ[rrow * 48 + part * 8]);
            asm volatile("global_load_async_to_lds_b128 %0, %1, off"
                         :: "v"(la), "v"(ga) : "memory");
            asm volatile("global_load_async_to_lds_b128 %0, %1, off"
                         :: "v"(lb), "v"(gb) : "memory");
        }
        asm volatile("s_wait_asynccnt 0x0" ::: "memory");

        // ---- fragments from LDS ----
        V16H af, bf;
        #pragma unroll
        for (int g = 0; g < 8; ++g) {
            const int ka = ((g & 3) * 2) + ((g >> 2) * 16) + lh * 8;
            const int kb = ((g & 3) * 2) + ((g >> 2) * 16) + lh * 8;
            af.u[g] = *reinterpret_cast<const unsigned*>(&sI[l16 * 48 + ka]);
            bf.u[g] = *reinterpret_cast<const unsigned*>(&sJ[l16 * 48 + kb]);
        }
        acc.v = WMMA_F16(af.v, bf.v, acc.v);
    }

    #pragma unroll
    for (int r = 0; r < 8; ++r) {
        const int i = i0 + lh * 8 + r;
        const int j = j0 + l16;
        KKT[((size_t)bh * 160 + i) * 160 + j] = (_Float16)(acc.f[r] * SCALE);
    }
}

// ---------------------------------------------------------------------------
// Kernel C: dots[b,h,i,j] = SCALE * sum_m q[b,m,i]*kkt[b,h,j,m]; softplus.
// One wave per 16(i) x 16(j) tile, K = 160 (5 steps of 32).
// ---------------------------------------------------------------------------
__global__ __launch_bounds__(32) void k_dots_softplus(
    const _Float16* __restrict__ Qf,   // (2,160,4096) f16
    const _Float16* __restrict__ KKT,  // (8,160,160) f16 (already *SCALE)
    float*          __restrict__ Out)  // flat (2,4,4096,160) f32
{
    const int bh = blockIdx.z;
    const int b  = bh >> 2;
    const int i0 = blockIdx.x * 16;    // D dimension
    const int j0 = blockIdx.y * 16;    // N dimension
    const int lane = threadIdx.x;
    const int l16 = lane & 15, lh = lane >> 4;

    const _Float16* qb = Qf  + (size_t)b  * 160 * 4096;
    const _Float16* kb = KKT + (size_t)bh * 160 * 160;

    V8F acc;
    #pragma unroll
    for (int r = 0; r < 8; ++r) acc.f[r] = 0.0f;

    for (int m0v = 0; m0v < 160; m0v += 32) {
        V16H af, bf;
        const int i = i0 + l16;
        #pragma unroll
        for (int g = 0; g < 8; ++g) {
            const int kb0 = ((g & 3) * 2) + ((g >> 2) * 16) + lh * 8 + m0v;
            // A[i, m] = q[b, m, i]: strided f16 gather (in-register transpose)
            af.h[2 * g]     = qb[(size_t)kb0       * 4096 + i];
            af.h[2 * g + 1] = qb[(size_t)(kb0 + 1) * 4096 + i];
            // B[m, j] = kkt row j, contiguous in m
            bf.u[g] = *reinterpret_cast<const unsigned*>(
                          kb + (size_t)(j0 + l16) * 160 + ((g & 3) * 2) + ((g >> 2) * 16) + lh * 8 + m0v);
        }
        acc.v = WMMA_F16(af.v, bf.v, acc.v);
    }

    #pragma unroll
    for (int r = 0; r < 8; ++r) {
        const int i = i0 + lh * 8 + r;
        const int j = j0 + l16;
        const float x = acc.f[r] * SCALE;
        const float sp = (x > 20.0f) ? x : log1pf(__expf(x));
        Out[((size_t)bh * 4096 + i) * 160 + j] = sp;
    }
}

// ---------------------------------------------------------------------------
extern "C" void kernel_launch(void* const* d_in, const int* in_sizes, int n_in,
                              void* d_out, int out_size, void* d_ws, size_t ws_size,
                              hipStream_t stream) {
    (void)in_sizes; (void)n_in; (void)out_size;

    const float* x  = (const float*)d_in[0];  // (2,160,4096)
    const float* y  = (const float*)d_in[1];  // (2,160,4096)
    const float* Wq = (const float*)d_in[2];  // (4096,4096)
    const float* Wk = (const float*)d_in[3];  // (4096,4096)
    float* out = (float*)d_out;

    // Workspace layout (f16 intermediates, ~5.5 MB total)
    _Float16* qf  = (_Float16*)d_ws;                   // 320*4096
    _Float16* kf  = qf + (size_t)320 * 4096;           // 320*4096
    _Float16* kkt = kf + (size_t)320 * 4096;           // 8*160*160
    const size_t need = ((size_t)320 * 4096 * 2 + (size_t)8 * 160 * 160) * sizeof(_Float16);
    if (ws_size < need) return;

    // q = y @ Wq^T ; k = x @ Wk^T   (f32 in, f16 out, f32 WMMA accumulate)
    dim3 gridA(320 / 64, 4096 / 128);
    k_linear_nt<<<gridA, 256, 0, stream>>>(y, Wq, qf, 320, 4096, 4096);
    k_linear_nt<<<gridA, 256, 0, stream>>>(x, Wk, kf, 320, 4096, 4096);

    // kkt = SCALE * k k^T per (b,h)
    k_kkt<<<dim3(10, 10, 8), 32, 0, stream>>>(kf, kkt);

    // dots = SCALE * q^T kkt^T per (b,h); softplus; flat store
    k_dots_softplus<<<dim3(256, 10, 8), 32, 0, stream>>>(qf, kkt, out);
}
